// IF4SR_53455162966094
// MI455X (gfx1250) — matrix operand order repaired and checked by reading, not compiled, not run on hardware
//
#include <hip/hip_runtime.h>
#include <hip/hip_bf16.h>

// ---------------------------------------------------------------------------
// Problem constants (match reference)
// ---------------------------------------------------------------------------
#define BB   512
#define LL   200
#define DD   256
#define NBLK 2
#define DHH  64
#define SCBH 512
#define FCBH 512
#define GNNL 2
#define HGAT 4
#define NI_N 100000
#define NT_N 20000
#define EI2T 300000
#define ET2T 60000

// ---------------------------------------------------------------------------
// WMMA GEMM: C[M,N] = op(A[M,K] @ B[K,N]) row-major packed, bf16 in, f32 acc.
// Requires K % 4 == 0 and N % 4 == 0 (true for every GEMM in this model).
// fuse: 0 = none, 1 = exact GELU, 2 = += R[M,N]
// Block: 128 threads = 4 wave32 waves (2x2); block tile 64x64; wave tile 32x32
// (2 A frags x 2 B frags -> 4 x v_wmma_f32_16x16x32_bf16 per K step).
// K loop split into full tiles (pointer-bump addressing, no clamps/selects)
// plus at most one masked tail tile. M/N edges handled purely by address
// clamping. Staging loads are forced into address space 1 so they lower to
// global_load_* (not flat_load_*, which would tick DScnt and contend with
// the ds_load_b128 fragment reads).
// ---------------------------------------------------------------------------
typedef __attribute__((ext_vector_type(16))) __bf16 v16bf;
typedef __attribute__((ext_vector_type(8)))  __bf16 v8bf;
typedef __attribute__((ext_vector_type(4)))  __bf16 v4bf;
typedef __attribute__((ext_vector_type(8)))  float  v8f;
typedef __attribute__((ext_vector_type(4)))  float  v4f;

typedef __attribute__((address_space(1))) const float* gcf1;
typedef __attribute__((address_space(1))) const v4f*   gcf4;
#define AS1F(p)  ((gcf1)(unsigned long long)(p))
#define AS1F4(p) ((gcf4)(unsigned long long)(p))

#define LDA_S 40
#define LDB_S 40

__device__ __forceinline__ float gelu_erf(float x) {
  return 0.5f * x * (1.0f + erff(x * 0.70710678118654752440f));
}

__global__ __launch_bounds__(128)
void wmma_gemm_kernel(const float* __restrict__ A, const float* __restrict__ Bm,
                      float* __restrict__ C, const float* __restrict__ R,
                      int M, int N, int K, int fuse)
{
  __shared__ __align__(16) __bf16 As[64 * LDA_S];
  __shared__ __align__(16) __bf16 Bs[64 * LDB_S];

  const int tid  = threadIdx.x;
  const int lane = tid & 31;
  const int w    = tid >> 5;
  const int wm   = w >> 1;      // 0..1 : 32 output rows
  const int wn   = w & 1;       // 0..1 : 32 output cols
  const int half = lane >> 4;   // 0/1
  const int r16  = lane & 15;
  const int m0 = blockIdx.y * 64;
  const int n0 = blockIdx.x * 64;

  // ---- per-thread staging geometry (loop-invariant, address-clamped) ----
  // A tile: 64 rows x 32 k = 512 float4 groups; 4 per thread
  int a_r[4], a_c4[4];
  const float* a_row[4];   // clamped row base (k=0)
  const float* aptr[4];    // running pointer: row + kk + c4
  #pragma unroll
  for (int i = 0; i < 4; i++) {
    int q = tid + i * 128;
    a_r[i]  = q >> 3;
    a_c4[i] = (q & 7) << 2;
    int gm  = m0 + a_r[i];
    a_row[i] = A + (size_t)(gm < M ? gm : (M - 1)) * K;
    aptr[i]  = a_row[i] + a_c4[i];
  }
  // B tile: 64 n x 8 k-quads = 512 groups; 4 per thread
  int b_n[4], b_kq[4];
  const float* b_col[4];   // clamped col base (k=0)
  const float* bptr[4];    // running pointer: col + (kk + kq)*N
  #pragma unroll
  for (int i = 0; i < 4; i++) {
    int q = tid + i * 128;
    b_n[i]  = q & 63;
    b_kq[i] = (q >> 6) << 2;
    int gn  = n0 + b_n[i];
    b_col[i] = Bm + (gn < N ? gn : (N - 1));
    bptr[i]  = b_col[i] + (size_t)b_kq[i] * N;
  }
  // prefetch pointer: one row per thread (clamped), next K slice
  const float* pfp;
  {
    int pr = m0 + (tid & 63);
    pfp = A + (size_t)(pr < M ? pr : (M - 1)) * K + 32;
  }

  v8f acc00, acc01, acc10, acc11;
  #pragma unroll
  for (int i = 0; i < 8; i++) { acc00[i]=0.f; acc01[i]=0.f; acc10[i]=0.f; acc11[i]=0.f; }

  // ---- loop-invariant LDS fragment pointers ----
  // A frag (16x32): lanes 0-15 M=r16; K = j + 16*half
  const __bf16* ap0 = &As[(wm * 32 + r16) * LDA_S];
  const __bf16* ap1 = ap0 + 16 * LDA_S;
  // B frags (32x16): lane n=r16 (+0 / +16 col subtiles); K = j + 16*half
  const __bf16* bp0 = &Bs[(wn * 32 + r16) * LDB_S + 16 * half];
  const __bf16* bp1 = bp0 + 16 * LDB_S;

  auto compute_tile = [&]() {
    v8bf a00 = *(const v8bf*)(ap0 + 16 * half);
    v8bf a01 = *(const v8bf*)(ap0 + 8 + 16 * half);
    v8bf a10 = *(const v8bf*)(ap1 + 16 * half);
    v8bf a11 = *(const v8bf*)(ap1 + 8 + 16 * half);
    v8bf b00 = *(const v8bf*)(bp0);
    v8bf b01 = *(const v8bf*)(bp0 + 8);
    v8bf b10 = *(const v8bf*)(bp1);
    v8bf b11 = *(const v8bf*)(bp1 + 8);
    v16bf av0, av1, bv0, bv1;
    #pragma unroll
    for (int i = 0; i < 8; i++) {
      av0[i] = a00[i]; av0[i + 8] = a01[i];
      av1[i] = a10[i]; av1[i + 8] = a11[i];
      bv0[i] = b00[i]; bv0[i + 8] = b01[i];
      bv1[i] = b10[i]; bv1[i + 8] = b11[i];
    }
    acc00 = __builtin_amdgcn_wmma_f32_16x16x32_bf16(false, av0, false, bv0, (short)0, acc00, false, false);
    acc01 = __builtin_amdgcn_wmma_f32_16x16x32_bf16(false, av0, false, bv1, (short)0, acc01, false, false);
    acc10 = __builtin_amdgcn_wmma_f32_16x16x32_bf16(false, av1, false, bv0, (short)0, acc10, false, false);
    acc11 = __builtin_amdgcn_wmma_f32_16x16x32_bf16(false, av1, false, bv1, (short)0, acc11, false, false);
  };

  // ================= main loop: full 32-wide K tiles =================
  for (int kk = 0; kk + 32 <= K; kk += 32) {
    v4f areg[4];
    #pragma unroll
    for (int i = 0; i < 4; i++) {
      areg[i] = *AS1F4(aptr[i]);
      aptr[i] += 32;
    }
    float breg[4][4];
    #pragma unroll
    for (int i = 0; i < 4; i++) {
      gcf1 bp = AS1F(bptr[i]);
      #pragma unroll
      for (int j = 0; j < 4; j++) breg[i][j] = bp[(size_t)j * N];
      bptr[i] += (size_t)32 * N;
    }
    if (kk + 64 <= K) { __builtin_prefetch(pfp, 0, 1); pfp += 32; }

    __syncthreads();   // prior iteration's LDS reads complete
    #pragma unroll
    for (int i = 0; i < 4; i++) {
      v4bf p;
      p[0] = (__bf16)areg[i][0]; p[1] = (__bf16)areg[i][1];
      p[2] = (__bf16)areg[i][2]; p[3] = (__bf16)areg[i][3];
      *(v4bf*)(&As[a_r[i] * LDA_S + a_c4[i]]) = p;
    }
    #pragma unroll
    for (int i = 0; i < 4; i++) {
      v4bf p;
      #pragma unroll
      for (int j = 0; j < 4; j++) p[j] = (__bf16)breg[i][j];
      *(v4bf*)(&Bs[b_n[i] * LDB_S + b_kq[i]]) = p;   // Bs[n][k] : K contiguous
    }
    __syncthreads();

    compute_tile();
  }

  // ================= tail tile (K % 32 != 0; runs at most once) ==========
  if (K & 31) {
    const int kfull = K & ~31;
    v4f areg[4];
    #pragma unroll
    for (int i = 0; i < 4; i++) {
      int gk = kfull + a_c4[i];
      bool ok = gk < K;                          // K%4==0 -> whole quad
      const float* p = ok ? aptr[i] : a_row[i];  // clamped, always valid
      v4f v = *AS1F4(p);
      if (!ok) { v[0] = 0.f; v[1] = 0.f; v[2] = 0.f; v[3] = 0.f; }
      areg[i] = v;
    }
    float breg[4][4];
    #pragma unroll
    for (int i = 0; i < 4; i++) {
      #pragma unroll
      for (int j = 0; j < 4; j++) {
        int gk = kfull + b_kq[i] + j;
        bool ok = gk < K;
        const float* p = ok ? (bptr[i] + (size_t)j * N) : b_col[i];
        float v = *AS1F(p);
        breg[i][j] = ok ? v : 0.0f;
      }
    }
    __syncthreads();
    #pragma unroll
    for (int i = 0; i < 4; i++) {
      v4bf p;
      p[0] = (__bf16)areg[i][0]; p[1] = (__bf16)areg[i][1];
      p[2] = (__bf16)areg[i][2]; p[3] = (__bf16)areg[i][3];
      *(v4bf*)(&As[a_r[i] * LDA_S + a_c4[i]]) = p;
    }
    #pragma unroll
    for (int i = 0; i < 4; i++) {
      v4bf p;
      #pragma unroll
      for (int j = 0; j < 4; j++) p[j] = (__bf16)breg[i][j];
      *(v4bf*)(&Bs[b_n[i] * LDB_S + b_kq[i]]) = p;
    }
    __syncthreads();

    compute_tile();
  }

  // ---- epilogue: C/D layout n = lane&15, acc[r] -> m = r + 8*half ----
  const int nnA = n0 + wn * 32 + r16;
  const int nnB = nnA + 16;
  #pragma unroll
  for (int rr = 0; rr < 8; rr++) {
    int mm0 = m0 + wm * 32 + rr + 8 * half;
    int mm1 = mm0 + 16;
    float v;
    if (mm0 < M && nnA < N) {
      v = acc00[rr];
      if (fuse == 1)      v = gelu_erf(v);
      else if (fuse == 2) v += R[(size_t)mm0 * N + nnA];
      C[(size_t)mm0 * N + nnA] = v;
    }
    if (mm0 < M && nnB < N) {
      v = acc01[rr];
      if (fuse == 1)      v = gelu_erf(v);
      else if (fuse == 2) v += R[(size_t)mm0 * N + nnB];
      C[(size_t)mm0 * N + nnB] = v;
    }
    if (mm1 < M && nnA < N) {
      v = acc10[rr];
      if (fuse == 1)      v = gelu_erf(v);
      else if (fuse == 2) v += R[(size_t)mm1 * N + nnA];
      C[(size_t)mm1 * N + nnA] = v;
    }
    if (mm1 < M && nnB < N) {
      v = acc11[rr];
      if (fuse == 1)      v = gelu_erf(v);
      else if (fuse == 2) v += R[(size_t)mm1 * N + nnB];
      C[(size_t)mm1 * N + nnB] = v;
    }
  }
}

// ---------------------------------------------------------------------------
// Row gather: out[r, 0:256] = tab[idx[r], 0:256]
// ---------------------------------------------------------------------------
__global__ void gather_kernel(const float* __restrict__ tab, const int* __restrict__ idx,
                              float* __restrict__ out, int rows)
{
  size_t i = (size_t)blockIdx.x * 256 + threadIdx.x;
  size_t total = (size_t)rows * 256;
  if (i >= total) return;
  size_t r = i >> 8;
  int c = (int)(i & 255);
  out[i] = tab[(size_t)idx[r] * 256 + c];
}

// ---------------------------------------------------------------------------
// LayerNorm over D=256; one wave per row
// ---------------------------------------------------------------------------
__global__ __launch_bounds__(256)
void ln_kernel(const float* __restrict__ x, const float* __restrict__ g,
               const float* __restrict__ b, float* __restrict__ y, int rows)
{
  int row  = (int)((blockIdx.x * 256 + threadIdx.x) >> 5);
  int lane = threadIdx.x & 31;
  if (row >= rows) return;
  const float* xr = x + (size_t)row * 256;
  float v[8];
  float s = 0.0f;
  #pragma unroll
  for (int j = 0; j < 8; j++) { v[j] = xr[lane + j * 32]; s += v[j]; }
  #pragma unroll
  for (int o = 16; o > 0; o >>= 1) s += __shfl_xor(s, o);
  float mu = s * (1.0f / 256.0f);
  float q = 0.0f;
  #pragma unroll
  for (int j = 0; j < 8; j++) { float d = v[j] - mu; q += d * d; }
  #pragma unroll
  for (int o = 16; o > 0; o >>= 1) q += __shfl_xor(q, o);
  float inv = 1.0f / sqrtf(q * (1.0f / 256.0f) + 1e-8f);
  float* yr = y + (size_t)row * 256;
  #pragma unroll
  for (int j = 0; j < 8; j++) {
    int d = lane + j * 32;
    yr[d] = (v[j] - mu) * inv * g[d] + b[d];
  }
}

// ---------------------------------------------------------------------------
// Batched 2D transpose: in [Bz,X,Y] -> out [Bz,Y,X]; out[b,y,x] = in[b,x,y] (+ res[b,y,x])
// ---------------------------------------------------------------------------
__global__ __launch_bounds__(256)
void transpose_kernel(const float* __restrict__ in, const float* __restrict__ res,
                      float* __restrict__ out, int X, int Y)
{
  __shared__ float tile[32][33];
  int b  = blockIdx.z;
  int x0 = blockIdx.x * 32, y0 = blockIdx.y * 32;
  int tx = threadIdx.x & 31, ty = threadIdx.x >> 5;
  size_t bo = (size_t)b * X * Y;
  const float* inb = in + bo;
  #pragma unroll
  for (int i = 0; i < 4; i++) {
    int x = x0 + ty + i * 8, y = y0 + tx;
    tile[ty + i * 8][tx] = (x < X && y < Y) ? inb[(size_t)x * Y + y] : 0.0f;
  }
  __syncthreads();
  float* outb = out + bo;
  #pragma unroll
  for (int i = 0; i < 4; i++) {
    int y = y0 + ty + i * 8, x = x0 + tx;
    if (x < X && y < Y) {
      float v = tile[tx][ty + i * 8];
      if (res) v += res[bo + (size_t)y * X + x];
      outb[(size_t)y * X + x] = v;
    }
  }
}

// ---------------------------------------------------------------------------
// Attention pooling over L: scores = V.wv, softmax over L, gi = sum alpha*V
// ---------------------------------------------------------------------------
__global__ __launch_bounds__(256)
void pool_kernel(const float* __restrict__ V, const float* __restrict__ wv,
                 float* __restrict__ gi)
{
  __shared__ float sc[256];
  __shared__ float red[256];
  int b = blockIdx.x, t = threadIdx.x;
  float s = -1e30f;
  if (t < LL) {
    const float* vr = V + ((size_t)b * LL + t) * 256;
    float a = 0.0f;
    for (int d = 0; d < 256; d++) a += vr[d] * wv[d];
    s = a; sc[t] = a;
  }
  red[t] = s;
  __syncthreads();
  for (int o = 128; o > 0; o >>= 1) { if (t < o) red[t] = fmaxf(red[t], red[t + o]); __syncthreads(); }
  float mx = red[0];
  __syncthreads();
  float e = (t < LL) ? expf(sc[t] - mx) : 0.0f;
  red[t] = e;
  __syncthreads();
  for (int o = 128; o > 0; o >>= 1) { if (t < o) red[t] += red[t + o]; __syncthreads(); }
  float inv = 1.0f / red[0];
  __syncthreads();
  if (t < LL) sc[t] = e * inv;
  __syncthreads();
  float g = 0.0f;
  for (int l = 0; l < LL; l++) g += sc[l] * V[((size_t)b * LL + l) * 256 + t];
  gi[(size_t)b * 256 + t] = g;
}

// ---------------------------------------------------------------------------
// pos/neg scoring: one wave per (b,l); logits[(b*L+l)*2 + {0,1}]
// ---------------------------------------------------------------------------
__global__ __launch_bounds__(256)
void score_kernel(const float* __restrict__ gi, const float* __restrict__ tab,
                  const int* __restrict__ pos, const int* __restrict__ neg,
                  float* __restrict__ out)
{
  int wid  = (int)((blockIdx.x * 256 + threadIdx.x) >> 5);
  int lane = threadIdx.x & 31;
  if (wid >= BB * LL) return;
  int b = wid / LL;
  const float* g = gi + (size_t)b * 256;
  float gv[8];
  #pragma unroll
  for (int j = 0; j < 8; j++) gv[j] = g[lane * 8 + j];
  const float* ep = tab + (size_t)pos[wid] * 256;
  const float* en = tab + (size_t)neg[wid] * 256;
  float sp = 0.0f, sn = 0.0f;
  #pragma unroll
  for (int j = 0; j < 8; j++) {
    sp += gv[j] * ep[lane * 8 + j];
    sn += gv[j] * en[lane * 8 + j];
  }
  #pragma unroll
  for (int o = 16; o > 0; o >>= 1) { sp += __shfl_xor(sp, o); sn += __shfl_xor(sn, o); }
  if (lane == 0) { out[(size_t)wid * 2] = sp; out[(size_t)wid * 2 + 1] = sn; }
}

// ---------------------------------------------------------------------------
// GAT helpers
// ---------------------------------------------------------------------------
__device__ __forceinline__ int   fkey(float f) { int b = __float_as_int(f); return b >= 0 ? b : (b ^ 0x7fffffff); }
__device__ __forceinline__ float funkey(int k) { int b = k >= 0 ? k : (k ^ 0x7fffffff); return __int_as_float(b); }
__device__ __forceinline__ float lrelu(float x) { return x > 0.0f ? x : 0.2f * x; }

__global__ void coef_kernel(const float* __restrict__ f, const float* __restrict__ a,
                            float* __restrict__ out, int N)
{
  int t = blockIdx.x * 256 + threadIdx.x;
  if (t >= N * 4) return;
  int n = t >> 2, h = t & 3;
  const float* fr = f + (size_t)n * 256 + h * 64;
  const float* ar = a + h * 64;
  float s = 0.0f;
  for (int d = 0; d < 64; d++) s += fr[d] * ar[d];
  out[t] = s;
}

__global__ void fill_f32_kernel(float* p, long n, float v)
{ long t = (long)blockIdx.x * 256 + threadIdx.x; if (t < n) p[t] = v; }
__global__ void fill_i32_kernel(int* p, long n, int v)
{ long t = (long)blockIdx.x * 256 + threadIdx.x; if (t < n) p[t] = v; }

__global__ void edge_max_kernel(const float* __restrict__ el, const float* __restrict__ er,
                                const int* __restrict__ src, const int* __restrict__ dst,
                                int* __restrict__ m, long E)
{
  long t = (long)blockIdx.x * 256 + threadIdx.x;
  if (t >= E * 4) return;
  long e = t >> 2; int h = (int)(t & 3);
  float v = lrelu(el[(size_t)src[e] * 4 + h] + er[(size_t)dst[e] * 4 + h]);
  atomicMax(&m[(size_t)dst[e] * 4 + h], fkey(v));
}

__global__ void edge_expsum_kernel(const float* __restrict__ el, const float* __restrict__ er,
                                   const int* __restrict__ src, const int* __restrict__ dst,
                                   const int* __restrict__ m, float* __restrict__ s, long E)
{
  long t = (long)blockIdx.x * 256 + threadIdx.x;
  if (t >= E * 4) return;
  long e = t >> 2; int h = (int)(t & 3);
  int d = dst[e];
  float v = lrelu(el[(size_t)src[e] * 4 + h] + er[(size_t)d * 4 + h]);
  atomicAdd(&s[(size_t)d * 4 + h], expf(v - funkey(m[(size_t)d * 4 + h])));
}

// one wave per edge: lane covers 8 consecutive dims -> single head per lane
__global__ __launch_bounds__(256)
void edge_accum_kernel(const float* __restrict__ el, const float* __restrict__ er,
                       const float* __restrict__ fs,
                       const int* __restrict__ src, const int* __restrict__ dst,
                       const int* __restrict__ m, const float* __restrict__ s,
                       float* __restrict__ out, long E)
{
  long w = ((long)blockIdx.x * 256 + threadIdx.x) >> 5;
  int lane = threadIdx.x & 31;
  if (w >= E) return;
  int sN = src[w], dN = dst[w];
  int h = lane >> 3;
  float v = lrelu(el[(size_t)sN * 4 + h] + er[(size_t)dN * 4 + h]);
  float alpha = expf(v - funkey(m[(size_t)dN * 4 + h])) / s[(size_t)dN * 4 + h];
  const float* fr = fs + (size_t)sN * 256 + lane * 8;
  float* orow = out + (size_t)dN * 256 + lane * 8;
  #pragma unroll
  for (int j = 0; j < 8; j++) atomicAdd(&orow[j], alpha * fr[j]);
}

__global__ void combine_kernel(const float* __restrict__ o1, const float* __restrict__ b1,
                               const float* __restrict__ o2, const float* __restrict__ b2,
                               float* __restrict__ out, int N)
{
  long t = (long)blockIdx.x * 256 + threadIdx.x;
  if (t >= (long)N * 256) return;
  int d = (int)(t & 255);
  out[t] = o1[t] + b1[d] + o2[t] + b2[d];
}

// ---------------------------------------------------------------------------
// Host orchestration
// ---------------------------------------------------------------------------
extern "C" void kernel_launch(void* const* d_in, const int* in_sizes, int n_in,
                              void* d_out, int out_size, void* d_ws, size_t ws_size,
                              hipStream_t stream)
{
  (void)in_sizes; (void)n_in; (void)out_size; (void)ws_size;

  const float* item_tab = (const float*)d_in[0];
  const float* tax_tab  = (const float*)d_in[1];
  const float* scb_ln_g = (const float*)d_in[2];
  const float* scb_ln_b = (const float*)d_in[3];
  const float* scb_w1   = (const float*)d_in[4];   // [NB,200,512]
  const float* scb_w2   = (const float*)d_in[5];   // [NB,512,200]
  const float* fcb_ln_g = (const float*)d_in[6];
  const float* fcb_ln_b = (const float*)d_in[7];
  const float* fcb_w1   = (const float*)d_in[8];   // [NB,64,512]
  const float* fcb_w2   = (const float*)d_in[9];   // [NB,512,64]
  const float* fcb_w3   = (const float*)d_in[10];  // [NB,256,256]
  const float* wv       = (const float*)d_in[11];
  const float* gat_w    = (const float*)d_in[12];  // [2,2,256,256]
  const float* gat_al   = (const float*)d_in[13];  // [2,2,4,64]
  const float* gat_ar   = (const float*)d_in[14];
  const float* gat_b    = (const float*)d_in[15];  // [2,2,256]
  const int* seq      = (const int*)d_in[16];
  const int* pos      = (const int*)d_in[17];
  const int* neg      = (const int*)d_in[18];
  const int* item_ids = (const int*)d_in[19];
  const int* tax_ids  = (const int*)d_in[20];
  const int* i2t_src  = (const int*)d_in[21];
  const int* i2t_dst  = (const int*)d_in[22];
  const int* t2t_src  = (const int*)d_in[23];
  const int* t2t_dst  = (const int*)d_in[24];
  float* out = (float*)d_out;

  float* base = (float*)d_ws;
  const size_t SZ_BLD = (size_t)BB * LL * DD;       // 26,214,400
  const size_t CHUNK  = 65536;                      // GEMM M-chunk
  float* V    = base;
  float* nV   = V    + SZ_BLD;
  float* nVs  = nV   + SZ_BLD;
  float* nVt  = nVs  + SZ_BLD;
  float* C2   = nVt  + SZ_BLD;                      // also reused as hout [409600,64]
  float* tmp  = C2   + SZ_BLD;                      // CHUNK x 512
  float* gi   = tmp  + CHUNK * 512;                 // [B,256]

  auto gemm = [&](const float* A, const float* Bm, float* C, const float* R,
                  int M, int N, int K, int fuse) {
    dim3 grid((N + 63) / 64, (M + 63) / 64);
    wmma_gemm_kernel<<<grid, dim3(128), 0, stream>>>(A, Bm, C, R, M, N, K, fuse);
  };

  // ================= global intention: MLP-mixer =================
  gather_kernel<<<(unsigned)(SZ_BLD / 256), 256, 0, stream>>>(item_tab, seq, V, BB * LL);

  for (int i = 0; i < NBLK; i++) {
    ln_kernel<<<BB * LL / 8, 256, 0, stream>>>(V, scb_ln_g + i * 256, scb_ln_b + i * 256, nV, BB * LL);
    transpose_kernel<<<dim3(7, 8, BB), 256, 0, stream>>>(nV, nullptr, nVt, LL, DD);
    for (int c = 0; c < 2; c++) {
      const float* Ac = nVt + (size_t)c * CHUNK * LL;
      gemm(Ac, scb_w1 + (size_t)i * LL * SCBH, tmp, nullptr, (int)CHUNK, SCBH, LL, 1);
      gemm(tmp, scb_w2 + (size_t)i * SCBH * LL, C2 + (size_t)c * CHUNK * LL, nullptr,
           (int)CHUNK, LL, SCBH, 0);
    }
    transpose_kernel<<<dim3(8, 7, BB), 256, 0, stream>>>(C2, nV, V, DD, LL);
    ln_kernel<<<BB * LL / 8, 256, 0, stream>>>(V, fcb_ln_g + i * 256, fcb_ln_b + i * 256, nVs, BB * LL);
    const int Mh = BB * LL * HGAT; // 409600
    for (int off = 0; off < Mh; off += (int)CHUNK) {
      int mc = (Mh - off < (int)CHUNK) ? (Mh - off) : (int)CHUNK;
      gemm(nVs + (size_t)off * DHH, fcb_w1 + (size_t)i * DHH * FCBH, tmp, nullptr, mc, FCBH, DHH, 1);
      gemm(tmp, fcb_w2 + (size_t)i * FCBH * DHH, C2 + (size_t)off * DHH, nullptr, mc, DHH, FCBH, 0);
    }
    gemm(C2, fcb_w3 + (size_t)i * DD * DD, V, nVs, BB * LL, DD, DD, 2);
  }

  pool_kernel<<<BB, 256, 0, stream>>>(V, wv, gi);
  score_kernel<<<(BB * LL * 32) / 256, 256, 0, stream>>>(gi, item_tab, pos, neg, out);

  // ================= local intention: hetero GAT =================
  float* hI    = base;                              // [NI,256]
  float* hTA   = hI    + (size_t)NI_N * 256;        // [NT,256]
  float* hTB   = hTA   + (size_t)NT_N * 256;
  float* fs_i  = hTB   + (size_t)NT_N * 256;        // [NI,256]
  float* fd0   = fs_i  + (size_t)NI_N * 256;        // [NT,256]
  float* ft    = fd0   + (size_t)NT_N * 256;        // [NT,256]
  float* el_i  = ft    + (size_t)NT_N * 256;        // [NI,4]
  float* er0   = el_i  + (size_t)NI_N * 4;          // [NT,4]
  float* el_t  = er0   + (size_t)NT_N * 4;
  float* er_t  = el_t  + (size_t)NT_N * 4;
  int*   m_a   = (int*)(er_t + (size_t)NT_N * 4);   // [NT,4]
  float* s_a   = (float*)m_a + (size_t)NT_N * 4;
  int*   m_b   = (int*)(s_a + (size_t)NT_N * 4);
  float* s_b   = (float*)m_b + (size_t)NT_N * 4;
  float* o_a   = s_b + (size_t)NT_N * 4;            // [NT,256]
  float* o_b   = o_a + (size_t)NT_N * 256;          // [NT,256]

  gather_kernel<<<NI_N, 256, 0, stream>>>(item_tab, item_ids, hI, NI_N);
  gather_kernel<<<NT_N, 256, 0, stream>>>(tax_tab, tax_ids, hTA, NT_N);

  const int KEY_NEG_INF = (int)0x807fffff;          // fkey(-inf)
  float* hin = hTA; float* hnx = hTB;
  for (int l = 0; l < GNNL; l++) {
    const float* W0 = gat_w + (size_t)(l * 2 + 0) * DD * DD;
    const float* W1 = gat_w + (size_t)(l * 2 + 1) * DD * DD;
    gemm(hI,  W0, fs_i, nullptr, NI_N, DD, DD, 0);
    gemm(hin, W0, fd0,  nullptr, NT_N, DD, DD, 0);
    gemm(hin, W1, ft,   nullptr, NT_N, DD, DD, 0);
    coef_kernel<<<(NI_N * 4 + 255) / 256, 256, 0, stream>>>(fs_i, gat_al + (l * 2 + 0) * 256, el_i, NI_N);
    coef_kernel<<<(NT_N * 4 + 255) / 256, 256, 0, stream>>>(fd0,  gat_ar + (l * 2 + 0) * 256, er0,  NT_N);
    coef_kernel<<<(NT_N * 4 + 255) / 256, 256, 0, stream>>>(ft,   gat_al + (l * 2 + 1) * 256, el_t, NT_N);
    coef_kernel<<<(NT_N * 4 + 255) / 256, 256, 0, stream>>>(ft,   gat_ar + (l * 2 + 1) * 256, er_t, NT_N);

    long n4 = (long)NT_N * 4, n256 = (long)NT_N * 256;
    fill_i32_kernel<<<(unsigned)((n4 + 255) / 256), 256, 0, stream>>>(m_a, n4, KEY_NEG_INF);
    fill_i32_kernel<<<(unsigned)((n4 + 255) / 256), 256, 0, stream>>>(m_b, n4, KEY_NEG_INF);
    fill_f32_kernel<<<(unsigned)((n4 + 255) / 256), 256, 0, stream>>>(s_a, n4, 0.0f);
    fill_f32_kernel<<<(unsigned)((n4 + 255) / 256), 256, 0, stream>>>(s_b, n4, 0.0f);
    fill_f32_kernel<<<(unsigned)((n256 + 255) / 256), 256, 0, stream>>>(o_a, n256, 0.0f);
    fill_f32_kernel<<<(unsigned)((n256 + 255) / 256), 256, 0, stream>>>(o_b, n256, 0.0f);

    edge_max_kernel<<<(EI2T * 4 + 255) / 256, 256, 0, stream>>>(el_i, er0, i2t_src, i2t_dst, m_a, EI2T);
    edge_expsum_kernel<<<(EI2T * 4 + 255) / 256, 256, 0, stream>>>(el_i, er0, i2t_src, i2t_dst, m_a, s_a, EI2T);
    edge_accum_kernel<<<(EI2T * 32 + 255) / 256, 256, 0, stream>>>(el_i, er0, fs_i, i2t_src, i2t_dst, m_a, s_a, o_a, EI2T);
    edge_max_kernel<<<(ET2T * 4 + 255) / 256, 256, 0, stream>>>(el_t, er_t, t2t_src, t2t_dst, m_b, ET2T);
    edge_expsum_kernel<<<(ET2T * 4 + 255) / 256, 256, 0, stream>>>(el_t, er_t, t2t_src, t2t_dst, m_b, s_b, ET2T);
    edge_accum_kernel<<<(ET2T * 32 + 255) / 256, 256, 0, stream>>>(el_t, er_t, ft, t2t_src, t2t_dst, m_b, s_b, o_b, ET2T);

    float* dest = (l == GNNL - 1) ? (out + (size_t)BB * LL * 2) : hnx;
    combine_kernel<<<(unsigned)((n256 + 255) / 256), 256, 0, stream>>>(
        o_a, gat_b + (l * 2 + 0) * 256, o_b, gat_b + (l * 2 + 1) * 256, dest, NT_N);
    float* t2 = hin; hin = hnx; hnx = t2;
  }
}
